// CaptioningRNN_60069412602579
// MI455X (gfx1250) — matrix operand order repaired and checked by reading, not compiled, not run on hardware
//
#include <hip/hip_runtime.h>
#include <stdint.h>

#define Nb    64
#define Tt    128
#define Dd    512
#define Hh    1024
#define K2    2560          // D + H + H
#define FOURH 4096

typedef __attribute__((ext_vector_type(16))) __bf16 v16bf;
typedef __attribute__((ext_vector_type(8)))  __bf16 v8bf;
typedef __attribute__((ext_vector_type(8)))  float  v8f;

__device__ __forceinline__ __bf16 f2bf(float f) {
    union { float f; uint32_t u; } v; v.f = f;
    uint32_t r = v.u + 0x7FFFu + ((v.u >> 16) & 1u);   // round-to-nearest-even
    uint16_t h = (uint16_t)(r >> 16);
    return __builtin_bit_cast(__bf16, h);
}

// Device-wide rendezvous: all `expected` workgroups arrive before any proceeds.
// Release-add, acquire-spin with s_sleep. Counters are re-zeroed every call.
__device__ __forceinline__ void grid_rendezvous(int* cnt, int expected) {
    __syncthreads();
    if (threadIdx.x == 0) {
        __threadfence();  // make this WG's abuf writes visible device-wide
        __hip_atomic_fetch_add(cnt, 1, __ATOMIC_RELEASE, __HIP_MEMORY_SCOPE_AGENT);
        while (__hip_atomic_load(cnt, __ATOMIC_ACQUIRE, __HIP_MEMORY_SCOPE_AGENT)
               < expected) {
            __builtin_amdgcn_s_sleep(1);
        }
        __threadfence();
    }
    __syncthreads();
}

// ---------------- Phase 0a: fuse+transpose weights into WT[ncol][k], bf16 ----
__global__ void __launch_bounds__(256)
wt_kernel(const float* __restrict__ Wx, const float* __restrict__ Wh,
          const float* __restrict__ Wattn, __bf16* __restrict__ WT) {
    int idx = blockIdx.x * 256 + threadIdx.x;       // covers 4096*2560 exactly
    int n = idx / K2;
    int k = idx - n * K2;
    float v;
    if (k < Dd)           v = Wx[(size_t)k * FOURH + n];
    else if (k < Dd + Hh) v = Wh[(size_t)(k - Dd) * FOURH + n];
    else                  v = Wattn[(size_t)(k - Dd - Hh) * FOURH + n];
    WT[idx] = f2bf(v);
}

// -------- Phase 0b: h0 = c0 = mean over 16 locations; zero step counters -----
__global__ void __launch_bounds__(256)
init_hc(const float* __restrict__ A, float* __restrict__ hbuf,
        float* __restrict__ cbuf, int* __restrict__ bar) {
    int idx = blockIdx.x * 256 + threadIdx.x;       // n*H + j, total 65536
    if (idx < Tt) bar[idx] = 0;
    const float* p = A + (size_t)idx * 16;
    float s = 0.f;
#pragma unroll
    for (int l = 0; l < 16; ++l) s += p[l];
    s *= (1.f / 16.f);
    hbuf[idx] = s;
    cbuf[idx] = s;
}

// ---------------- Per-step fused kernel ---------------------------------------
// Phase A: WG i packs abuf row n=i = bf16[xt | h | attn] (attention incl.)
// Rendezvous (all 64 WGs).
// Phase B: WMMA GEMM a = abuf(64x2560) @ WT^T for 16 h-columns x 4 gates,
//          then LSTM gates + h/c update, out write.
__global__ void __launch_bounds__(512)
step_kernel(const float* __restrict__ x, const float* __restrict__ Afull,
            const __bf16* __restrict__ WT, const float* __restrict__ b,
            __bf16* __restrict__ abuf, float* __restrict__ hbuf,
            float* __restrict__ cbuf, float* __restrict__ out,
            int* __restrict__ bar, int t) {
    __shared__ float aLDS[4][64][16];   // [gate][m][jj]
    __shared__ float part[32][16];      // [sub][l]
    __shared__ float sc[16];

    int tid = threadIdx.x;

    // ======== Phase A: attention + pack for batch row n = blockIdx.x ========
    {
        int n = blockIdx.x;
        const float* Af = Afull + (size_t)n * (Hh * 16);
        const float* hp = hbuf + (size_t)n * Hh;

        // deterministic reduction: scores[l] = dot(h, Af[:,l]) / sqrt(H)
        int l = tid & 15, sub = tid >> 4;        // 32 subs x 16 l
        float ps = 0.f;
        int hbase = sub * 32;
#pragma unroll 4
        for (int i = 0; i < 32; ++i) {
            float hv = hp[hbase + i];
            ps += hv * Af[(size_t)(hbase + i) * 16 + l];
        }
        part[sub][l] = ps;
        __syncthreads();
        if (tid < 16) {
            float s = 0.f;
#pragma unroll
            for (int i = 0; i < 32; ++i) s += part[i][tid];
            sc[tid] = s * 0.03125f;              // 1/sqrt(1024)
        }
        __syncthreads();

        // softmax weights, computed redundantly per thread (deterministic)
        float w[16];
        float mx = sc[0];
#pragma unroll
        for (int i = 1; i < 16; ++i) mx = fmaxf(mx, sc[i]);
        float sum = 0.f;
#pragma unroll
        for (int i = 0; i < 16; ++i) { w[i] = __expf(sc[i] - mx); sum += w[i]; }
        float inv = 1.f / sum;
#pragma unroll
        for (int i = 0; i < 16; ++i) w[i] *= inv;

        __bf16* arow = abuf + (size_t)n * K2;
        const float* xp = x + ((size_t)n * Tt + t) * Dd;
        for (int k = tid; k < Dd; k += 512) arow[k] = f2bf(xp[k]);
        for (int j = tid; j < Hh; j += 512) {
            arow[Dd + j] = f2bf(hp[j]);
            float av = 0.f;
#pragma unroll
            for (int i = 0; i < 16; ++i) av += Af[(size_t)j * 16 + i] * w[i];
            arow[Dd + Hh + j] = f2bf(av);
        }
    }

    // ======== all 64 WGs must finish packing abuf before any GEMM reads =====
    grid_rendezvous(bar + t, Nb);

    // ======== Phase B: WMMA GEMM + gates ====================================
    int wv   = tid >> 5;
    int lane = tid & 31;
    int gate = wv & 3;
    int mt   = wv >> 2;
    int j0   = blockIdx.x * 16;

    int nn = lane & 15;     // B column / A row-in-tile / C column
    int kh = lane >> 4;     // K-half selector

    // B: lane holds 16 consecutive K of column (gate*H + j0 + nn)
    const __bf16* bp = WT + (size_t)(gate * Hh + j0 + nn) * K2 + kh * 16;
    // A: lane holds row (16*mt + nn); K = k0+16*kh..+15 (contiguous)
    const __bf16* ap = abuf + (size_t)(mt * 16 + nn) * K2 + kh * 16;

    // Two independent accumulators -> two independent WMMA dep-chains
    v8f acc0 = {}, acc1 = {};
    for (int k0 = 0; k0 < K2; k0 += 64) {
        {
            v16bf bfrag = *(const v16bf*)(bp + k0);
            v8bf  lo    = *(const v8bf*)(ap + k0);
            v8bf  hi    = *(const v8bf*)(ap + k0 + 8);
            v16bf afrag = __builtin_shufflevector(lo, hi,
                            0, 1, 2, 3, 4, 5, 6, 7, 8, 9, 10, 11, 12, 13, 14, 15);
            acc0 = __builtin_amdgcn_wmma_f32_16x16x32_bf16(
                       false, afrag, false, bfrag, (short)0, acc0, false, false);
        }
        {
            v16bf bfrag = *(const v16bf*)(bp + k0 + 32);
            v8bf  lo    = *(const v8bf*)(ap + k0 + 32);
            v8bf  hi    = *(const v8bf*)(ap + k0 + 40);
            v16bf afrag = __builtin_shufflevector(lo, hi,
                            0, 1, 2, 3, 4, 5, 6, 7, 8, 9, 10, 11, 12, 13, 14, 15);
            acc1 = __builtin_amdgcn_wmma_f32_16x16x32_bf16(
                       false, afrag, false, bfrag, (short)0, acc1, false, false);
        }
    }
    v8f acc = acc0 + acc1;

    // C layout: VGPR r, lanes 0-15 -> M=r, lanes 16-31 -> M=8+r; N = lane&15
#pragma unroll
    for (int r = 0; r < 8; ++r)
        aLDS[gate][mt * 16 + kh * 8 + r][nn] = acc[r];
    __syncthreads();

    // Gates: 1024 (m, jj) outputs over 512 threads
    for (int e = tid; e < Nb * 16; e += 512) {
        int m  = e >> 4;
        int jj = e & 15;
        int j  = j0 + jj;
        float ai = aLDS[0][m][jj] + b[j];
        float af = aLDS[1][m][jj] + b[Hh + j];
        float ao = aLDS[2][m][jj] + b[2 * Hh + j];
        float ag = aLDS[3][m][jj] + b[3 * Hh + j];
        float ig = 1.f / (1.f + __expf(-ai));
        float fg = 1.f / (1.f + __expf(-af));
        float og = 1.f / (1.f + __expf(-ao));
        float gg = tanhf(ag);
        int off  = m * Hh + j;
        float cn = fg * cbuf[off] + ig * gg;
        float hn = og * tanhf(cn);
        cbuf[off] = cn;
        hbuf[off] = hn;
        out[((size_t)m * Tt + t) * Hh + j] = hn;
    }
}

extern "C" void kernel_launch(void* const* d_in, const int* in_sizes, int n_in,
                              void* d_out, int out_size, void* d_ws, size_t ws_size,
                              hipStream_t stream) {
    (void)in_sizes; (void)n_in; (void)out_size; (void)ws_size;
    const float* x     = (const float*)d_in[0];
    const float* A     = (const float*)d_in[1];
    const float* Wx    = (const float*)d_in[2];
    const float* Wh    = (const float*)d_in[3];
    const float* Wattn = (const float*)d_in[4];
    const float* b     = (const float*)d_in[5];
    float* out = (float*)d_out;

    char* ws = (char*)d_ws;
    __bf16* WT   = (__bf16*)(ws);                                  // 20,971,520 B
    __bf16* abuf = (__bf16*)(ws + 20971520);                       //    327,680 B
    float*  hbuf = (float*)(ws + 20971520 + 327680);               //    262,144 B
    float*  cbuf = (float*)(ws + 20971520 + 327680 + 262144);      //    262,144 B
    int*    bar  = (int*)  (ws + 20971520 + 327680 + 524288);      //        512 B

    // Phase 0: weight fuse/transpose to bf16 (L2-resident thereafter); h0/c0;
    // zero the per-step rendezvous counters (runs every call => graph-safe).
    wt_kernel<<<(FOURH * K2) / 256, 256, 0, stream>>>(Wx, Wh, Wattn, WT);
    init_hc<<<(Nb * Hh) / 256, 256, 0, stream>>>(A, hbuf, cbuf, bar);

    // Serial recurrence: one fused kernel per step (attention pack -> global
    // rendezvous -> WMMA GEMM + gates). 64 WGs x 512 threads: co-resident on
    // any MI455X-class part, so the rendezvous is safe.
    for (int t = 0; t < Tt; ++t) {
        step_kernel<<<Nb, 512, 0, stream>>>(x, A, WT, b, abuf, hbuf, cbuf, out,
                                            bar, t);
    }
}
